// Attention_19207093747743
// MI455X (gfx1250) — matrix operand order repaired and checked by reading, not compiled, not run on hardware
//
#include <hip/hip_runtime.h>

// ---------------------------------------------------------------------------
// CDNA5 (gfx1250) fused talking-heads attention block.
// bf16 WMMA pipeline + Tensor Data Mover (TDM) LDS staging of shared weights.
// ---------------------------------------------------------------------------

typedef __attribute__((ext_vector_type(16))) __bf16 v16bf;
typedef __attribute__((ext_vector_type(8)))  float  v8f;
typedef __attribute__((ext_vector_type(4)))  unsigned int u32x4;
typedef __attribute__((ext_vector_type(8)))  int  i32x8;
typedef __attribute__((ext_vector_type(4)))  int  i32x4;

namespace {
constexpr int TOKENS  = 8;
constexpr int NHEADS  = 16;
constexpr int DIM     = 1024;
constexpr int HID     = 4096;
constexpr int OUTD    = 1024;
constexpr int NQKV    = DIM + DIM + HID;        // 6144 (Q|K|V fused)
constexpr int BATCH   = 2048;
constexpr int MTOK    = BATCH * TOKENS;         // 16384 token rows
constexpr float ATTN_SCALE = 0.125f;            // 64^-0.5
constexpr float LNEPS = 1e-5f;

constexpr int KT1 = DIM / 32;                   // 32 k-tiles (QKV gemm)
constexpr int KT3 = HID / 32;                   // 128 k-tiles (out gemm)

constexpr int CHUNK_KT   = 4;                   // k-tiles per TDM chunk
constexpr int FRAG_ELEMS = 512;                 // bf16 per (tile,kt) fragment chunk
constexpr int PANEL_ELEMS = CHUNK_KT * FRAG_ELEMS;     // 2048 bf16 per n-tile per chunk
constexpr int LDSB_ELEMS  = 4 * PANEL_ELEMS;           // 4 n-tiles -> 16 KB buffer

// workspace layout (bytes)
constexpr size_t OFF_XP    = 0;                                   // 32 MB  bf16 packed X
constexpr size_t OFF_WQKVP = OFF_XP    + (size_t)MTOK * DIM  * 2; // 12 MB  bf16 packed Wq|Wk|Wv
constexpr size_t OFF_WPP   = OFF_WQKVP + (size_t)DIM  * NQKV * 2; // 8 MB   bf16 packed Wp
constexpr size_t OFF_QKV   = OFF_WPP   + (size_t)HID  * OUTD * 2; // 192 MB bf16 QKV rows
constexpr size_t OFF_AP    = OFF_QKV   + (size_t)MTOK * NQKV * 2; // 128 MB bf16 packed activations
}

__device__ __forceinline__ v8f wmma_bf16(v16bf a, v16bf b, v8f c) {
  return __builtin_amdgcn_wmma_f32_16x16x32_bf16(
      /*neg_a=*/false, a, /*neg_b=*/false, b,
      /*c_mod=*/(short)0, c, /*reuse_a=*/false, /*reuse_b=*/false);
}

// ---------------------------------------------------------------------------
// TDM: DMA a 2D set of weight-fragment panels (nrows x row_elems bf16, row
// stride row_stride_elems) from global into LDS.  D# per CDNA5 ISA ch.8:
//   group0: count=1 | lds_addr | global_addr[56:0] | type=2
//   group1: data_size=2B, tensor_dim0/1, tile_dim0/1, tensor_dim0_stride
// Issued once per wave (EXEC ignored); tracked by TENSORcnt.
// ---------------------------------------------------------------------------
__device__ __forceinline__ void tdm_load_panels(const __bf16* gsrc, __bf16* ldst,
                                                unsigned row_elems, unsigned nrows,
                                                unsigned row_stride_elems) {
  unsigned long long ga = (unsigned long long)gsrc;
  unsigned lds_off = (unsigned)(unsigned long long)ldst;  // flat addr[31:0] == LDS offset

  u32x4 G0;
  G0[0] = 1u;                                             // count=1 (user descriptor)
  G0[1] = lds_off;                                        // lds_addr (bytes)
  G0[2] = (unsigned)ga;                                   // global_addr[31:0]
  G0[3] = (unsigned)((ga >> 32) & 0x01FFFFFFu) | (2u << 30); // addr[56:32] | type=2

  i32x8 G1;
  G1[0] = (int)(1u << 16);                                // data_size=1 -> 2-byte elems
  G1[1] = (int)((row_elems & 0xFFFFu) << 16);             // tensor_dim0[15:0]  (bits 63:48)
  G1[2] = (int)(((row_elems >> 16) & 0xFFFFu)             // tensor_dim0[31:16] (bits 79:64)
              | ((nrows & 0xFFFFu) << 16));               // tensor_dim1[15:0]  (bits 95:80)
  G1[3] = (int)(((nrows >> 16) & 0xFFFFu)                 // tensor_dim1[31:16]
              | ((row_elems & 0xFFFFu) << 16));           // tile_dim0 (bits 127:112)
  G1[4] = (int)(nrows & 0xFFFFu);                         // tile_dim1 (tile_dim2 = 0)
  G1[5] = (int)row_stride_elems;                          // tensor_dim0_stride[31:0]
  G1[6] = 0;                                              // stride[47:32] | dim1_stride lo
  G1[7] = 0;

  i32x4 Z4 = {0, 0, 0, 0};
#if defined(__clang_major__) && (__clang_major__ >= 23)
  i32x8 Z8 = {0, 0, 0, 0, 0, 0, 0, 0};
  __builtin_amdgcn_tensor_load_to_lds(G0, G1, Z4, Z4, Z8, 0);
#else
  __builtin_amdgcn_tensor_load_to_lds(G0, G1, Z4, Z4, 0);
#endif
}

// ---------------------------------------------------------------------------
// Pack fp32 activations X[MTOK,DIM] into WMMA A-fragment order (bf16).
// ---------------------------------------------------------------------------
__global__ __launch_bounds__(256) void pack_x_kernel(const float* __restrict__ X,
                                                     __bf16* __restrict__ Xp) {
  int id   = blockIdx.x * 256 + threadIdx.x;     // 1,048,576 chunks
  int lane = id & 31;
  int kt   = (id >> 5) % KT1;
  int mt   = (id >> 5) / KT1;
  int m    = mt * 16 + (lane & 15);
  int kb   = (lane >= 16) ? 8 : 0;
  const float* src = X + (size_t)m * DIM + kt * 32;
  __bf16* dst = Xp + (size_t)id * 16;
#pragma unroll
  for (int j = 0; j < 8; ++j) dst[j]     = (__bf16)src[kb + j];
#pragma unroll
  for (int j = 0; j < 8; ++j) dst[8 + j] = (__bf16)src[kb + 16 + j];
}

// ---------------------------------------------------------------------------
// Pack Wq|Wk|Wv ([in,out] row-major) into WMMA B-fragment order.
// ---------------------------------------------------------------------------
__global__ __launch_bounds__(256) void pack_wqkv_kernel(const float* __restrict__ Wq,
                                                        const float* __restrict__ Wk,
                                                        const float* __restrict__ Wv,
                                                        __bf16* __restrict__ Wqkvp) {
  int id   = blockIdx.x * 256 + threadIdx.x;     // 393,216 chunks
  int lane = id & 31;
  int kt   = (id >> 5) % KT1;
  int nt   = (id >> 5) / KT1;
  int n    = nt * 16 + (lane & 15);
  int kb   = (lane >= 16) ? 8 : 0;
  const float* src; int col, ncols;
  if (n < DIM)            { src = Wq; col = n;           ncols = DIM; }
  else if (n < 2 * DIM)   { src = Wk; col = n - DIM;     ncols = DIM; }
  else                    { src = Wv; col = n - 2 * DIM; ncols = HID; }
  __bf16* dst = Wqkvp + (size_t)id * 16;
#pragma unroll
  for (int j = 0; j < 16; ++j)
    dst[j] = (__bf16)src[(size_t)(kt * 32 + kb + ((j < 8) ? j : (j + 8))) * ncols + col];
}

__global__ __launch_bounds__(256) void pack_wp_kernel(const float* __restrict__ Wp,
                                                      __bf16* __restrict__ Wpp) {
  int id   = blockIdx.x * 256 + threadIdx.x;     // 262,144 chunks
  int lane = id & 31;
  int kt   = (id >> 5) % KT3;
  int nt   = (id >> 5) / KT3;
  int n    = nt * 16 + (lane & 15);
  int kb   = (lane >= 16) ? 8 : 0;
  __bf16* dst = Wpp + (size_t)id * 16;
#pragma unroll
  for (int j = 0; j < 16; ++j)
    dst[j] = (__bf16)Wp[(size_t)(kt * 32 + kb + ((j < 8) ? j : (j + 8))) * OUTD + n];
}

// ---------------------------------------------------------------------------
// QKV GEMM: [MTOK,DIM] x [DIM,NQKV] -> bf16 QKV rows (+bias).
// 8 waves/block share one nt-group: TDM double-buffers the B panels in LDS.
// ---------------------------------------------------------------------------
__global__ __launch_bounds__(256) void qkv_gemm_kernel(const __bf16* __restrict__ Xp,
                                                       const __bf16* __restrict__ Wqkvp,
                                                       const float* __restrict__ bq,
                                                       const float* __restrict__ bk,
                                                       const float* __restrict__ bv,
                                                       __bf16* __restrict__ QKV) {
  __shared__ __align__(32) __bf16 sB[2][LDSB_ELEMS];   // 2 x 16 KB weight panels

  const int wid  = threadIdx.x >> 5;
  const int lane = threadIdx.x & 31;
  const int wave = blockIdx.x * 8 + wid;
  const int gm = wave & 511;                      // 512 groups of 2 m-tiles
  const int gn = wave >> 9;                       // 96 groups of 4 n-tiles
  const int mt0 = gm * 2;
  const int nt0 = gn * 4;

  const v16bf* Ap = (const v16bf*)Xp;
  const __bf16* Bpanel = Wqkvp + (size_t)nt0 * KT1 * FRAG_ELEMS;

  constexpr int NCH = KT1 / CHUNK_KT;             // 8 chunks
  if (wid == 0)
    tdm_load_panels(Bpanel, &sB[0][0], PANEL_ELEMS, 4, KT1 * FRAG_ELEMS);

  v8f acc[2][4] = {};
  for (int c = 0; c < NCH; ++c) {
    if (wid == 0) {
      if (c + 1 < NCH) {
        tdm_load_panels(Bpanel + (size_t)(c + 1) * CHUNK_KT * FRAG_ELEMS,
                        &sB[(c + 1) & 1][0], PANEL_ELEMS, 4, KT1 * FRAG_ELEMS);
        __builtin_amdgcn_s_wait_tensorcnt(1);     // oldest DMA (current chunk) done
      } else {
        __builtin_amdgcn_s_wait_tensorcnt(0);
      }
    }
    __syncthreads();
    const v16bf* Bl = (const v16bf*)&sB[c & 1][0];
#pragma unroll
    for (int k = 0; k < CHUNK_KT; ++k) {
      const int kt = c * CHUNK_KT + k;
      const int abase = (mt0 * KT1 + kt) * 32 + lane;
      __builtin_prefetch(&Ap[abase + 8 * 32], 0, 1);   // global_prefetch_b8 (A stream)
      v16bf a0 = Ap[abase];
      v16bf a1 = Ap[abase + KT1 * 32];
      v16bf b0 = Bl[(0 * CHUNK_KT + k) * 32 + lane];
      v16bf b1 = Bl[(1 * CHUNK_KT + k) * 32 + lane];
      v16bf b2 = Bl[(2 * CHUNK_KT + k) * 32 + lane];
      v16bf b3 = Bl[(3 * CHUNK_KT + k) * 32 + lane];
      acc[0][0] = wmma_bf16(a0, b0, acc[0][0]);
      acc[0][1] = wmma_bf16(a0, b1, acc[0][1]);
      acc[0][2] = wmma_bf16(a0, b2, acc[0][2]);
      acc[0][3] = wmma_bf16(a0, b3, acc[0][3]);
      acc[1][0] = wmma_bf16(a1, b0, acc[1][0]);
      acc[1][1] = wmma_bf16(a1, b1, acc[1][1]);
      acc[1][2] = wmma_bf16(a1, b2, acc[1][2]);
      acc[1][3] = wmma_bf16(a1, b3, acc[1][3]);
    }
    __syncthreads();                              // buffer free for chunk c+2
  }

  const int ncol = lane & 15;
  const int rbase = (lane >> 4) * 8;              // C frag: rows r / r+8
#pragma unroll
  for (int j = 0; j < 4; ++j) {
    int n = (nt0 + j) * 16 + ncol;
    float bias = (n < DIM) ? bq[n] : ((n < 2 * DIM) ? bk[n - DIM] : bv[n - 2 * DIM]);
#pragma unroll
    for (int i = 0; i < 2; ++i) {
      int row0 = (mt0 + i) * 16 + rbase;
#pragma unroll
      for (int r = 0; r < 8; ++r)
        QKV[(size_t)(row0 + r) * NQKV + n] = (__bf16)(acc[i][j][r] + bias);
    }
  }
}

// ---------------------------------------------------------------------------
// Per-batch-row attention + talking heads + softmax + attn*V + LayerNorm +
// swish, emitting activations pre-packed in WMMA A-fragment order.
// ---------------------------------------------------------------------------
__global__ __launch_bounds__(256) void attn_kernel(const __bf16* __restrict__ QKV,
                                                   const float* __restrict__ Wl,
                                                   const float* __restrict__ bl,
                                                   const float* __restrict__ Ww,
                                                   const float* __restrict__ bw,
                                                   const float* __restrict__ gamma,
                                                   const float* __restrict__ beta,
                                                   __bf16* __restrict__ Apk) {
  __shared__ __align__(16) __bf16 sQK[TOKENS][2 * DIM];   // 32 KB  (Q|K per token)
  __shared__ float sA[NHEADS][TOKENS][TOKENS];            // 4 KB
  __shared__ float sB[NHEADS][TOKENS][TOKENS];            // 4 KB
  __shared__ float sW[2][NHEADS * NHEADS];                // 2 KB
  __shared__ float sO[HID];                               // 16 KB
  __shared__ float sRed[18];

  const int b   = blockIdx.x;
  const int tid = threadIdx.x;
  const __bf16* base = QKV + (size_t)b * TOKENS * NQKV;

  sW[0][tid] = Wl[tid];
  sW[1][tid] = Ww[tid];
#pragma unroll
  for (int t = 0; t < TOKENS; ++t) {
    const uint4* s = (const uint4*)(base + (size_t)t * NQKV);
    uint4* d = (uint4*)sQK[t];
    d[tid] = s[tid];                                      // 256 x 16B = 4 KB/row
  }
  __syncthreads();

  // attn logits: q . k * scale
  for (int e = tid; e < NHEADS * TOKENS * TOKENS; e += 256) {
    int j = e & 7, i = (e >> 3) & 7, h = e >> 6;
    const __bf16* q = &sQK[i][h * 64];
    const __bf16* k = &sQK[j][DIM + h * 64];
    float acc = 0.f;
#pragma unroll
    for (int d = 0; d < 64; ++d) acc += (float)q[d] * (float)k[d];
    sA[h][i][j] = acc * ATTN_SCALE;
  }
  __syncthreads();

  // talking heads (pre-softmax)
  for (int e = tid; e < NHEADS * TOKENS * TOKENS; e += 256) {
    int j = e & 7, i = (e >> 3) & 7, g = e >> 6;
    float acc = bl[g];
#pragma unroll
    for (int h = 0; h < NHEADS; ++h) acc += sA[h][i][j] * sW[0][h * 16 + g];
    sB[g][i][j] = acc;
  }
  __syncthreads();

  // softmax over last axis, per (g,i)
  if (tid < NHEADS * TOKENS) {
    int i = tid & 7, g = tid >> 3;
    float mx = -1e30f;
#pragma unroll
    for (int j = 0; j < TOKENS; ++j) mx = fmaxf(mx, sB[g][i][j]);
    float s = 0.f;
#pragma unroll
    for (int j = 0; j < TOKENS; ++j) { float ev = __expf(sB[g][i][j] - mx); sB[g][i][j] = ev; s += ev; }
    float inv = 1.f / s;
#pragma unroll
    for (int j = 0; j < TOKENS; ++j) sB[g][i][j] *= inv;
  }
  __syncthreads();

  // talking heads (post-softmax)
  for (int e = tid; e < NHEADS * TOKENS * TOKENS; e += 256) {
    int j = e & 7, i = (e >> 3) & 7, g = e >> 6;
    float acc = bw[g];
#pragma unroll
    for (int h = 0; h < NHEADS; ++h) acc += sB[h][i][j] * sW[1][h * 16 + g];
    sA[g][i][j] = acc;
  }
  __syncthreads();

  const int mt = b >> 1;                     // packed m-tile (16 rows = 2 batch rows)
  for (int t = 0; t < TOKENS; ++t) {
    for (int c = tid; c < HID; c += 256) {
      int h = c >> 8;
      float acc = 0.f;
#pragma unroll
      for (int j = 0; j < TOKENS; ++j)
        acc += sA[h][t][j] * (float)base[(size_t)j * NQKV + 2 * DIM + c];
      sO[c] = acc;
    }
    __syncthreads();

    // LayerNorm reduction (wave32 shuffle + LDS)
    float s = 0.f, s2 = 0.f;
    for (int c = tid; c < HID; c += 256) { float v = sO[c]; s += v; s2 += v * v; }
    for (int off = 16; off > 0; off >>= 1) {
      s  += __shfl_down(s,  off, 32);
      s2 += __shfl_down(s2, off, 32);
    }
    if ((tid & 31) == 0) { sRed[tid >> 5] = s; sRed[8 + (tid >> 5)] = s2; }
    __syncthreads();
    if (tid == 0) {
      float ts = 0.f, ts2 = 0.f;
#pragma unroll
      for (int w = 0; w < 8; ++w) { ts += sRed[w]; ts2 += sRed[8 + w]; }
      float mu = ts * (1.f / HID);
      float var = ts2 * (1.f / HID) - mu * mu;
      sRed[16] = mu;
      sRed[17] = rsqrtf(var + LNEPS);
    }
    __syncthreads();
    float mu = sRed[16], rstd = sRed[17];

    for (int c = tid; c < HID; c += 256) {
      float v = (sO[c] - mu) * rstd * gamma[c] + beta[c];
      sO[c] = v / (1.f + __expf(-v));
    }
    __syncthreads();

    // write token row in WMMA A-fragment order (32B contiguous per thread)
    {
      int mrow = (b & 1) * 8 + t;
      int kt   = tid >> 1;                   // 0..127
      int half = tid & 1;
      int lane = mrow + 16 * half;
      __bf16* dst = Apk + ((size_t)(mt * KT3 + kt) * 32 + lane) * 16;
#pragma unroll
      for (int j = 0; j < 16; ++j) {
        int kk = half ? ((j < 8) ? (8 + j) : (16 + j))
                      : ((j < 8) ? j : (8 + j));
        dst[j] = (__bf16)sO[kt * 32 + kk];
      }
    }
    __syncthreads();
  }
}

// ---------------------------------------------------------------------------
// Output GEMM: [MTOK,HID] x [HID,OUTD] + bp -> fp32 d_out, TDM-staged B.
// ---------------------------------------------------------------------------
__global__ __launch_bounds__(256) void out_gemm_kernel(const __bf16* __restrict__ Apk,
                                                       const __bf16* __restrict__ Wpp,
                                                       const float* __restrict__ bp,
                                                       float* __restrict__ out) {
  __shared__ __align__(32) __bf16 sB[2][LDSB_ELEMS];   // 2 x 16 KB weight panels

  const int wid  = threadIdx.x >> 5;
  const int lane = threadIdx.x & 31;
  const int wave = blockIdx.x * 8 + wid;
  const int gm = wave & 511;
  const int gn = wave >> 9;                       // 16 groups of 4 n-tiles
  const int mt0 = gm * 2;
  const int nt0 = gn * 4;

  const v16bf* Ap = (const v16bf*)Apk;
  const __bf16* Bpanel = Wpp + (size_t)nt0 * KT3 * FRAG_ELEMS;

  constexpr int NCH = KT3 / CHUNK_KT;             // 32 chunks
  if (wid == 0)
    tdm_load_panels(Bpanel, &sB[0][0], PANEL_ELEMS, 4, KT3 * FRAG_ELEMS);

  v8f acc[2][4] = {};
  for (int c = 0; c < NCH; ++c) {
    if (wid == 0) {
      if (c + 1 < NCH) {
        tdm_load_panels(Bpanel + (size_t)(c + 1) * CHUNK_KT * FRAG_ELEMS,
                        &sB[(c + 1) & 1][0], PANEL_ELEMS, 4, KT3 * FRAG_ELEMS);
        __builtin_amdgcn_s_wait_tensorcnt(1);
      } else {
        __builtin_amdgcn_s_wait_tensorcnt(0);
      }
    }
    __syncthreads();
    const v16bf* Bl = (const v16bf*)&sB[c & 1][0];
#pragma unroll
    for (int k = 0; k < CHUNK_KT; ++k) {
      const int kt = c * CHUNK_KT + k;
      const int abase = (mt0 * KT3 + kt) * 32 + lane;
      __builtin_prefetch(&Ap[abase + 8 * 32], 0, 1);
      v16bf a0 = Ap[abase];
      v16bf a1 = Ap[abase + KT3 * 32];
      v16bf b0 = Bl[(0 * CHUNK_KT + k) * 32 + lane];
      v16bf b1 = Bl[(1 * CHUNK_KT + k) * 32 + lane];
      v16bf b2 = Bl[(2 * CHUNK_KT + k) * 32 + lane];
      v16bf b3 = Bl[(3 * CHUNK_KT + k) * 32 + lane];
      acc[0][0] = wmma_bf16(a0, b0, acc[0][0]);
      acc[0][1] = wmma_bf16(a0, b1, acc[0][1]);
      acc[0][2] = wmma_bf16(a0, b2, acc[0][2]);
      acc[0][3] = wmma_bf16(a0, b3, acc[0][3]);
      acc[1][0] = wmma_bf16(a1, b0, acc[1][0]);
      acc[1][1] = wmma_bf16(a1, b1, acc[1][1]);
      acc[1][2] = wmma_bf16(a1, b2, acc[1][2]);
      acc[1][3] = wmma_bf16(a1, b3, acc[1][3]);
    }
    __syncthreads();
  }

  const int ncol = lane & 15;
  const int rbase = (lane >> 4) * 8;
#pragma unroll
  for (int j = 0; j < 4; ++j) {
    int n = (nt0 + j) * 16 + ncol;
    float bias = bp[n];
#pragma unroll
    for (int i = 0; i < 2; ++i) {
      int row0 = (mt0 + i) * 16 + rbase;
#pragma unroll
      for (int r = 0; r < 8; ++r)
        out[(size_t)(row0 + r) * OUTD + n] = acc[i][j][r] + bias;
    }
  }
}

// ---------------------------------------------------------------------------
extern "C" void kernel_launch(void* const* d_in, const int* in_sizes, int n_in,
                              void* d_out, int out_size, void* d_ws, size_t ws_size,
                              hipStream_t stream) {
  (void)in_sizes; (void)n_in; (void)out_size; (void)ws_size;

  const float* x     = (const float*)d_in[0];
  const float* Wq    = (const float*)d_in[1];
  const float* bq    = (const float*)d_in[2];
  const float* Wk    = (const float*)d_in[3];
  const float* bk    = (const float*)d_in[4];
  const float* Wv    = (const float*)d_in[5];
  const float* bv    = (const float*)d_in[6];
  const float* Wl    = (const float*)d_in[7];
  const float* bl    = (const float*)d_in[8];
  const float* Ww    = (const float*)d_in[9];
  const float* bw    = (const float*)d_in[10];
  const float* gamma = (const float*)d_in[11];
  const float* beta  = (const float*)d_in[12];
  const float* Wp    = (const float*)d_in[13];
  const float* bp    = (const float*)d_in[14];

  char* ws = (char*)d_ws;
  __bf16* Xp    = (__bf16*)(ws + OFF_XP);
  __bf16* Wqkvp = (__bf16*)(ws + OFF_WQKVP);
  __bf16* Wpp   = (__bf16*)(ws + OFF_WPP);
  __bf16* QKV   = (__bf16*)(ws + OFF_QKV);
  __bf16* Apk   = (__bf16*)(ws + OFF_AP);

  // pack activations + weights to bf16 WMMA fragment order
  pack_x_kernel   <<<4096, 256, 0, stream>>>(x, Xp);
  pack_wqkv_kernel<<<1536, 256, 0, stream>>>(Wq, Wk, Wv, Wqkvp);
  pack_wp_kernel  <<<1024, 256, 0, stream>>>(Wp, Wpp);

  // fused Q|K|V projection: 49152 waves = 6144 blocks x 8 waves
  qkv_gemm_kernel<<<6144, 256, 0, stream>>>(Xp, Wqkvp, bq, bk, bv, QKV);

  // attention + talking heads + softmax + LN + swish, one block per batch row
  attn_kernel<<<BATCH, 256, 0, stream>>>(QKV, Wl, bl, Ww, bw, gamma, beta, Apk);

  // output projection: 8192 waves = 1024 blocks x 8 waves
  out_gemm_kernel<<<1024, 256, 0, stream>>>(Apk, Wpp, bp, (float*)d_out);
}
